// CausalSelfAttention_7550552507095
// MI455X (gfx1250) — matrix operand order repaired and checked by reading, not compiled, not run on hardware
//
#include <hip/hip_runtime.h>
#include <hip/hip_bf16.h>

#define B_    2
#define T_    2048
#define C_    1024
#define H_    16
#define HD_   64
#define M4    (B_*T_)    // 4096 tokens
#define K_    C_         // 1024
#define N_QKV (3*C_)     // 3072
#define NKSTEP (K_/32)   // 32 K-steps of 32

typedef __attribute__((ext_vector_type(16))) __bf16 bf16x16;
typedef __attribute__((ext_vector_type(8)))  float  f32x8;
typedef int v4i_vs __attribute__((vector_size(16)));   // matches builtin param type

#define GLOBAL_AS __attribute__((address_space(1)))
#define LDS_AS    __attribute__((address_space(3)))

#if __has_builtin(__builtin_amdgcn_global_load_async_to_lds_b128)
#define HAS_ASYNC_LDS 1
#else
#define HAS_ASYNC_LDS 0
#endif

__device__ inline f32x8 wmma_bf16(bf16x16 a, bf16x16 b, f32x8 c) {
  // D = A(16x32 bf16) x B(32x16 bf16) + C(16x16 f32)
  return __builtin_amdgcn_wmma_f32_16x16x32_bf16(false, a, false, b, (short)0, c, false, false);
}

// async 16B global -> LDS copy (falls back to sync copy if builtin missing)
__device__ inline void async_cp16(const void* g, void* l) {
#if HAS_ASYNC_LDS
  __builtin_amdgcn_global_load_async_to_lds_b128(
      (GLOBAL_AS v4i_vs*)g, (LDS_AS v4i_vs*)l, 0, 0);
#else
  *(uint4*)l = *(const uint4*)g;
#endif
}

__device__ inline void wait_async0() {
#if __has_builtin(__builtin_amdgcn_s_wait_asynccnt)
  __builtin_amdgcn_s_wait_asynccnt(0);
#else
  asm volatile("s_wait_asynccnt 0x0" ::: "memory");
#endif
}

// 32-byte fragment load, requires 16B alignment only.
__device__ inline bf16x16 load16(const __bf16* p) {
  union { uint4 u[2]; bf16x16 v; } t;
  const uint4* q = (const uint4*)p;
  t.u[0] = q[0]; t.u[1] = q[1];
  return t.v;
}

__device__ inline float rmax16(float v) {
  #pragma unroll
  for (int off = 1; off < 16; off <<= 1) v = fmaxf(v, __shfl_xor(v, off, 32));
  return v;
}
__device__ inline float rsum16(float v) {
  #pragma unroll
  for (int off = 1; off < 16; off <<= 1) v += __shfl_xor(v, off, 32);
  return v;
}

// ---------------- fp32 -> bf16 convert (linear) ----------------
__global__ void cvt_kernel(const float* __restrict__ src, __bf16* __restrict__ dst, int n4) {
  int i = blockIdx.x * blockDim.x + threadIdx.x;
  if (i < n4) {
    float4 f = ((const float4*)src)[i];
    union { __bf16 h[4]; uint2 u; } o;
    o.h[0] = (__bf16)f.x; o.h[1] = (__bf16)f.y;
    o.h[2] = (__bf16)f.z; o.h[3] = (__bf16)f.w;
    ((uint2*)dst)[i] = o.u;
  }
}

// ---------------- fp32 [R,Ccols] -> bf16 transposed [Ccols,R] ----------------
__global__ void cvtT_kernel(const float* __restrict__ src, __bf16* __restrict__ dst,
                            int R, int Ccols) {
  __shared__ float tile[32][33];
  const int c0 = blockIdx.x * 32, r0 = blockIdx.y * 32;
  const int tx = threadIdx.x, ty = threadIdx.y;   // (32, 8)
  #pragma unroll
  for (int i = 0; i < 32; i += 8)
    tile[ty + i][tx] = src[(size_t)(r0 + ty + i) * Ccols + c0 + tx];
  __syncthreads();
  #pragma unroll
  for (int i = 0; i < 32; i += 8)
    dst[(size_t)(c0 + ty + i) * R + r0 + tx] = (__bf16)tile[tx][ty + i];
}

// ---------------- async double-buffered bf16 GEMM ----------------
// D[M,N] = A[M,K] * Bn[K,N] + bias.  Block 256 thr = 8 waves, tile 128x128,
// wave tile 32x64 (8 WMMA per K-step of 32).  Tiles move via async global->LDS.
// mode 0: fp32 D (out projection).  mode 1: scatter qkv -> q / kT / v bf16.
__launch_bounds__(256)
__global__ void gemm_kernel(const __bf16* __restrict__ A, const __bf16* __restrict__ Bn,
                            const float* __restrict__ bias, int N, int mode,
                            float* __restrict__ Dout,
                            __bf16* __restrict__ qb, __bf16* __restrict__ kTb,
                            __bf16* __restrict__ vb) {
  __shared__ __bf16 As[2][128 * 40];   // 128 m x 32 k, stride 40 (80B, 16B-mult)
  __shared__ __bf16 Bs[2][32 * 136];   // 32 k x 128 n, stride 136 (272B, 16B-mult)
  const int tid  = threadIdx.x;
  const int lane = tid & 31, wid = tid >> 5;
  const int wm = wid & 3, wn = wid >> 2;         // 4x2 wave grid
  const int hl = lane >> 4, ln = lane & 15;
  const int m0 = blockIdx.x * 128;
  const int n0 = blockIdx.y * 128;

  // coop copy slots: A = 512 16B-chunks (2/thread), B = 512 16B-chunks (2/thread)
  const int arow = tid >> 2, acol = (tid & 3) * 8;
  const int brow = tid >> 4, bcol = (tid & 15) * 8;

  auto stage = [&](int kt, int buf) {
    const int k0 = kt * 32;
    async_cp16(A + (size_t)(m0 + arow) * K_ + k0 + acol,      &As[buf][arow * 40 + acol]);
    async_cp16(A + (size_t)(m0 + arow + 64) * K_ + k0 + acol, &As[buf][(arow + 64) * 40 + acol]);
    async_cp16(Bn + (size_t)(k0 + brow) * N + n0 + bcol,      &Bs[buf][brow * 136 + bcol]);
    async_cp16(Bn + (size_t)(k0 + brow + 16) * N + n0 + bcol, &Bs[buf][(brow + 16) * 136 + bcol]);
  };

  f32x8 c[2][4] = {};
  stage(0, 0);
  for (int kt = 0; kt < NKSTEP; ++kt) {
    const int cur = kt & 1;
    wait_async0();          // my chunks of buf[cur] arrived
    __syncthreads();        // everyone's chunks arrived; buf[cur^1] free for overwrite
    if (kt + 1 < NKSTEP) stage(kt + 1, cur ^ 1);   // overlap copy with math

    bf16x16 a[2], b[4];
    #pragma unroll
    for (int mi = 0; mi < 2; ++mi) {
      const __bf16* row = &As[cur][(wm * 32 + mi * 16 + ln) * 40];
      #pragma unroll
      for (int j = 0; j < 8; ++j) {               // A 16x32 layout: K pairs, +8*half
        int kk = (j < 4 ? 2 * j : 16 + 2 * (j - 4)) + 8 * hl;
        a[mi][2 * j]     = row[kk];
        a[mi][2 * j + 1] = row[kk + 1];
      }
    }
    #pragma unroll
    for (int ni = 0; ni < 4; ++ni)                // B 32x16 layout: lane = n, k per element (table order)
      #pragma unroll
      for (int s = 0; s < 16; ++s)
        b[ni][s] = Bs[cur][((s < 8 ? s : s + 8) + 8 * hl) * 136 + wn * 64 + ni * 16 + ln];
    #pragma unroll
    for (int mi = 0; mi < 2; ++mi)
      #pragma unroll
      for (int ni = 0; ni < 4; ++ni)
        c[mi][ni] = wmma_bf16(a[mi], b[ni], c[mi][ni]);
  }

  if (mode == 0) {
    #pragma unroll
    for (int mi = 0; mi < 2; ++mi)
      #pragma unroll
      for (int ni = 0; ni < 4; ++ni)
        #pragma unroll
        for (int g = 0; g < 8; ++g) {
          const int row = m0 + wm * 32 + mi * 16 + g + 8 * hl;
          const int col = n0 + wn * 64 + ni * 16 + ln;
          Dout[(size_t)row * N + col] = c[mi][ni][g] + bias[col];
        }
  } else {
    const int sec = n0 >> 10;                     // uniform: 128 | 1024
    __bf16* dst = (sec == 0) ? qb : (sec == 1 ? kTb : vb);
    #pragma unroll
    for (int mi = 0; mi < 2; ++mi)
      #pragma unroll
      for (int ni = 0; ni < 4; ++ni)
        #pragma unroll
        for (int g = 0; g < 8; ++g) {
          const int row = m0 + wm * 32 + mi * 16 + g + 8 * hl;
          const int col = n0 + wn * 64 + ni * 16 + ln;
          const int cs = col & 1023, hh = cs >> 6, d = cs & 63;
          const int bb = row >> 11, t = row & 2047;
          const int bh = bb * H_ + hh;
          const size_t idx = (sec == 1) ? ((size_t)bh * HD_ + d) * T_ + t   // K transposed
                                        : ((size_t)bh * T_ + t) * HD_ + d;
          dst[idx] = (__bf16)(c[mi][ni][g] + bias[col]);
        }
  }
}

// ---------------- Flash attention: 1 wave = 16 q-rows, 32 k-rows per step ----
__launch_bounds__(128)
__global__ void attn_kernel(const __bf16* __restrict__ qb, const __bf16* __restrict__ kTb,
                            const __bf16* __restrict__ vb, __bf16* __restrict__ yatt) {
  __shared__ __bf16 P[4][16 * 32];   // per-wave P staging (C-frag -> A-frag)
  const int tid  = threadIdx.x;
  const int lane = tid & 31, wid = tid >> 5;
  const int hl = lane >> 4, ln = lane & 15;
  const int bh = blockIdx.y;
  const int bb = bh >> 4, hh = bh & 15;
  const int qbase = blockIdx.x * 64 + wid * 16;
  __bf16* Pw = P[wid];

  bf16x16 q0, q1;                                    // Q A-frags, d=0..31 / 32..63
  {
    const __bf16* qrow = qb + ((size_t)bh * T_ + qbase + ln) * HD_;
    #pragma unroll
    for (int j = 0; j < 8; ++j) {
      int kk = (j < 4 ? 2 * j : 16 + 2 * (j - 4)) + 8 * hl;
      q0[2 * j] = qrow[kk];      q0[2 * j + 1] = qrow[kk + 1];
      q1[2 * j] = qrow[kk + 32]; q1[2 * j + 1] = qrow[kk + 33];
    }
  }
  float mi[8], li[8];
  f32x8 o[4] = {};
  #pragma unroll
  for (int g = 0; g < 8; ++g) { mi[g] = -1e30f; li[g] = 0.f; }

  const int nPairs = (qbase >> 5) + 1;               // causal bound
  const __bf16* kb0 = kTb + (size_t)bh * HD_ * T_;
  const __bf16* vb0 = vb  + (size_t)bh * T_ * HD_;

  for (int it = 0; it < nPairs; ++it) {
    const int t0 = it * 32, t1 = t0 + 16;
    if (it + 1 < nPairs) {                           // warm L2/WGP$ for next step
      __builtin_prefetch(kb0 + (size_t)lane * T_ + t0 + 32, 0, 0);
      __builtin_prefetch(kb0 + (size_t)(32 + lane) * T_ + t0 + 32, 0, 0);
      __builtin_prefetch(vb0 + (size_t)(t0 + 32 + lane) * HD_, 0, 0);
    }
    f32x8 s0 = {}, s1 = {};
    // S = Q * K^T : kT rows are d (k per element, table order), lane = t column.
    bf16x16 k00, k01, k10, k11;
    #pragma unroll
    for (int s = 0; s < 16; ++s) {
      const int dk = (s < 8 ? s : s + 8) + 8 * hl;
      k00[s] = kb0[(size_t)dk * T_ + t0 + ln];
      k01[s] = kb0[(size_t)(32 + dk) * T_ + t0 + ln];
      k10[s] = kb0[(size_t)dk * T_ + t1 + ln];
      k11[s] = kb0[(size_t)(32 + dk) * T_ + t1 + ln];
    }
    s0 = wmma_bf16(q0, k00, s0);
    s0 = wmma_bf16(q1, k01, s0);
    s1 = wmma_bf16(q0, k10, s1);
    s1 = wmma_bf16(q1, k11, s1);

    float alpha[8];
    #pragma unroll
    for (int g = 0; g < 8; ++g) {
      const int qrow = qbase + g + 8 * hl;
      float v0 = (t0 + ln <= qrow) ? s0[g] * 0.125f : -1e30f;   // 1/sqrt(64)
      float v1 = (t1 + ln <= qrow) ? s1[g] * 0.125f : -1e30f;
      float mnew = fmaxf(mi[g], rmax16(fmaxf(v0, v1)));
      alpha[g] = __expf(mi[g] - mnew);
      mi[g] = mnew;
      float p0 = __expf(v0 - mnew);
      float p1 = __expf(v1 - mnew);
      li[g] = li[g] * alpha[g] + rsum16(p0 + p1);
      Pw[(g + 8 * hl) * 32 + ln]      = (__bf16)p0;
      Pw[(g + 8 * hl) * 32 + 16 + ln] = (__bf16)p1;
    }
    #pragma unroll
    for (int c4 = 0; c4 < 4; ++c4)
      #pragma unroll
      for (int g = 0; g < 8; ++g) o[c4][g] *= alpha[g];

    asm volatile("s_wait_dscnt 0" ::: "memory");     // P writes -> A-frag reads
    bf16x16 pa;
    #pragma unroll
    for (int j = 0; j < 8; ++j) {
      int kk = (j < 4 ? 2 * j : 16 + 2 * (j - 4)) + 8 * hl;
      pa[2 * j]     = Pw[ln * 32 + kk];
      pa[2 * j + 1] = Pw[ln * 32 + kk + 1];
    }
    #pragma unroll
    for (int c4 = 0; c4 < 4; ++c4) {
      bf16x16 vf;
      #pragma unroll
      for (int s = 0; s < 16; ++s) {
        const int trow = min(t0 + (s < 8 ? s : s + 8) + 8 * hl, T_ - 1);   // clamp fully-masked tail
        vf[s] = vb0[(size_t)trow * HD_ + c4 * 16 + ln];
      }
      o[c4] = wmma_bf16(pa, vf, o[c4]);
    }
    asm volatile("s_wait_dscnt 0" ::: "memory");     // reads done before next writes
  }

  #pragma unroll
  for (int c4 = 0; c4 < 4; ++c4)
    #pragma unroll
    for (int g = 0; g < 8; ++g) {
      int t = qbase + g + 8 * hl;
      float v = o[c4][g] / li[g];
      yatt[((size_t)bb * T_ + t) * C_ + hh * HD_ + c4 * 16 + ln] = (__bf16)v;
    }
}

// ---------------- host launcher ----------------
extern "C" void kernel_launch(void* const* d_in, const int* in_sizes, int n_in,
                              void* d_out, int out_size, void* d_ws, size_t ws_size,
                              hipStream_t stream) {
  (void)in_sizes; (void)n_in; (void)out_size; (void)ws_size;
  const float* x     = (const float*)d_in[0];
  const float* Wattn = (const float*)d_in[1];
  const float* battn = (const float*)d_in[2];
  const float* Wproj = (const float*)d_in[3];
  const float* bproj = (const float*)d_in[4];
  float* out = (float*)d_out;

  char* w = (char*)d_ws;
  const size_t MB = 1024 * 1024;
  __bf16* xbf  = (__bf16*)(w);             // 8 MB  x bf16 [M4,K]
  __bf16* waT  = (__bf16*)(w + 8 * MB);    // 6 MB  W_attn^T bf16 [K,3C]
  __bf16* wpT  = (__bf16*)(w + 14 * MB);   // 2 MB  W_proj^T bf16 [K,C]
  __bf16* qb   = (__bf16*)(w + 16 * MB);   // 8 MB  q  [B,H,T,HD]
  __bf16* kTb  = (__bf16*)(w + 24 * MB);   // 8 MB  kT [B,H,HD,T]
  __bf16* vbuf = (__bf16*)(w + 32 * MB);   // 8 MB  v  [B,H,T,HD]
  __bf16* yatt = (__bf16*)(w + 40 * MB);   // 8 MB  attn out bf16 [B*T, C]

  int n = M4 * K_ / 4;
  cvt_kernel<<<(n + 255) / 256, 256, 0, stream>>>(x, xbf, n);
  cvtT_kernel<<<dim3(K_ / 32, N_QKV / 32), dim3(32, 8), 0, stream>>>(Wattn, waT, N_QKV, K_);
  cvtT_kernel<<<dim3(K_ / 32, C_ / 32),    dim3(32, 8), 0, stream>>>(Wproj, wpT, C_, K_);

  gemm_kernel<<<dim3(M4 / 128, N_QKV / 128), 256, 0, stream>>>(
      xbf, waT, battn, N_QKV, 1, nullptr, qb, kTb, vbuf);

  attn_kernel<<<dim3(T_ / 64, B_ * H_), 128, 0, stream>>>(qb, kTb, vbuf, yatt);

  gemm_kernel<<<dim3(M4 / 128, C_ / 128), 256, 0, stream>>>(
      yatt, wpT, bproj, C_, 0, out, nullptr, nullptr, nullptr);
}
